// ODE_RNN_12575664243120
// MI455X (gfx1250) — matrix operand order repaired and checked by reading, not compiled, not run on hardware
//
#include <hip/hip_runtime.h>
#include <hip/hip_bf16.h>
#include <math.h>

// ---------------------------------------------------------------------------
// ODE-RNN fused kernel for MI455X (gfx1250, wave32, WMMA).
//  - 64 workgroups x 256 threads (8 waves); each WG owns 16 batch rows and
//    runs the entire T=128 sequential recurrence with LDS-resident state.
//  - All GEMMs via v_wmma_f32_16x16x32_bf16 (bf16 in, f32 accumulate).
//  - Weights converted once to bf16 into workspace (L2-resident thereafter).
// ---------------------------------------------------------------------------

typedef __attribute__((ext_vector_type(16))) __bf16 v16bf;
typedef __attribute__((ext_vector_type(8)))  __bf16 v8bf;
typedef __attribute__((ext_vector_type(8)))  float  v8f;

#define BATCH   1024
#define T_STEPS 128
#define HID     256
#define ROWS    16        // batch rows per workgroup (WMMA M)
#define LDA     264       // bf16 staging row stride (pad: 132 dwords % 64 = 4)
#define LDF     260       // f32 buffer row stride   (260 dwords % 64 = 4)
#define LDG     772       // GRU gate buffer stride  (772 dwords % 64 = 4)

// ---- A-operand fragment: 16x32 bf16 tile from LDS, contiguous k order (k = 16*half + s) ----
// lanes 0-15: elems 0..15 = K(k0..k0+15),    row M=lane
// lanes16-31: elems 0..15 = K(k0+16..k0+31), row M=lane-16
__device__ __forceinline__ v16bf load_a_frag(const __bf16* base, int m, int half, int k0) {
  const __bf16* p = base + m * LDA + k0 + half * 16;
  v8bf lo = *(const v8bf*)p;
  v8bf hi = *(const v8bf*)(p + 8);
  v16bf r;
#pragma unroll
  for (int j = 0; j < 8; ++j) { r[j] = lo[j]; r[j + 8] = hi[j]; }
  return r;
}

// ---- one 16xN GEMM slice: NCT 16x16 output tiles per wave, K = 256 --------
// B = W^T, so B column n == W row n (contiguous 32B per k-chunk per lane).
template <int NCT>
__device__ __forceinline__ void gemm_k256(const __bf16* sAbase, const __bf16* __restrict__ W,
                                          int wave, int lane, v8f* acc) {
  const int half = lane >> 4;
  const int m    = lane & 15;            // == B-operand column within tile
  const v8f z = {0.f, 0.f, 0.f, 0.f, 0.f, 0.f, 0.f, 0.f};
#pragma unroll
  for (int c = 0; c < NCT; ++c) acc[c] = z;
#pragma unroll
  for (int kk = 0; kk < HID; kk += 32) {
    v16bf a = load_a_frag(sAbase, m, half, kk);
#pragma unroll
    for (int c = 0; c < NCT; ++c) {
      const int col = (wave * NCT + c) * 16 + m;
      v16bf b = *(const v16bf*)(W + col * HID + kk + half * 16);
      acc[c] = __builtin_amdgcn_wmma_f32_16x16x32_bf16(
          /*neg_a=*/false, a, /*neg_b=*/false, b,
          /*c_mod=*/(short)0, acc[c], /*reuse_a=*/false, /*reuse_b=*/false);
    }
  }
}

__global__ __launch_bounds__(256) void ode_rnn_fused(
    const float* __restrict__ data, const float* __restrict__ t,
    const float* __restrict__ b1,   const float* __restrict__ b2,
    const float* __restrict__ W_ih, const float* __restrict__ b_ih,
    const float* __restrict__ b_hh,
    const float* __restrict__ bo1,  const float* __restrict__ Wo2,
    const float* __restrict__ bo2,  const float* __restrict__ h0,
    const __bf16* __restrict__ wW1, const __bf16* __restrict__ wW2,
    const __bf16* __restrict__ wWhh, const __bf16* __restrict__ wWo1,
    float* __restrict__ out, float* __restrict__ hpf)
{
  __shared__ float  sH[ROWS][LDF];     // f32 hidden state (persists across steps)
  __shared__ float  sAcc[ROWS][LDF];   // RK4 k-accumulator / head scratch
  __shared__ float  sGh[ROWS][LDG];    // GRU gh gates (768 wide)
  __shared__ __bf16 sA[ROWS][LDA];     // bf16 A-operand staging
  __shared__ __bf16 sB[ROWS][LDA];     // bf16 inner-layer staging
  __shared__ float  sRed[ROWS][17];    // head dot-product partials

  const int tid  = threadIdx.x;
  const int wave = tid >> 5;
  const int lane = tid & 31;
  const int half = lane >> 4;
  const int ln16 = lane & 15;
  const int r0   = blockIdx.x * ROWS;

  for (int idx = tid; idx < ROWS * HID; idx += 256) {
    const int row = idx >> 8, j = idx & 255;
    sH[row][j] = h0[(r0 + row) * HID + j];
  }
  __syncthreads();

  float tprev = 0.f;
  for (int step = 0; step < T_STEPS; ++step) {
    const float tcur = t[step];          // row 0 of t, uniform across batch
    const float dt = tcur - tprev;
    tprev = tcur;
    __builtin_prefetch(&data[step * BATCH + r0], 0, 0);

    // stage y0 = h into bf16 A operand
    for (int idx = tid; idx < ROWS * HID; idx += 256) {
      const int row = idx >> 8, j = idx & 255;
      sA[row][j] = (__bf16)sH[row][j];
    }
    __syncthreads();

    // ---------------- RK4 over [tprev, tcur]: 4 evals of ode_f -------------
#pragma unroll
    for (int stage = 0; stage < 4; ++stage) {
      v8f acc[2];
      // u = y @ W1^T + b1  -> bf16 sB
      gemm_k256<2>(&sA[0][0], wW1, wave, lane, acc);
#pragma unroll
      for (int c = 0; c < 2; ++c) {
        const int col = (wave * 2 + c) * 16 + ln16;
        const float bb = b1[col];
#pragma unroll
        for (int v = 0; v < 8; ++v)
          sB[v + half * 8][col] = (__bf16)(acc[c][v] + bb);
      }
      __syncthreads();

      // k = tanh(u @ W2^T + b2); RK4 combine + next stage input
      gemm_k256<2>(&sB[0][0], wW2, wave, lane, acc);
#pragma unroll
      for (int c = 0; c < 2; ++c) {
        const int col = (wave * 2 + c) * 16 + ln16;
        const float bb = b2[col];
#pragma unroll
        for (int v = 0; v < 8; ++v) {
          const int row = v + half * 8;
          const float k = tanhf(acc[c][v] + bb);
          if (stage == 0) {
            sAcc[row][col] = k;
            sA[row][col] = (__bf16)(sH[row][col] + 0.5f * dt * k);
          } else if (stage == 1) {
            sAcc[row][col] += 2.f * k;
            sA[row][col] = (__bf16)(sH[row][col] + 0.5f * dt * k);
          } else if (stage == 2) {
            sAcc[row][col] += 2.f * k;
            sA[row][col] = (__bf16)(sH[row][col] + dt * k);
          } else {
            const float hp = sH[row][col] + (dt * (1.f / 6.f)) * (sAcc[row][col] + k);
            sH[row][col] = hp;                 // h' (= hp, pre-GRU)
            sA[row][col] = (__bf16)hp;         // A operand for head / GRU
          }
        }
      }
      __syncthreads();
    }

    // ---------------- head: tanh((hp @ Wo1^T + bo1) . Wo2 + bo2) -----------
    {
      v8f acc[2];
      gemm_k256<2>(&sA[0][0], wWo1, wave, lane, acc);
#pragma unroll
      for (int c = 0; c < 2; ++c) {
        const int col = (wave * 2 + c) * 16 + ln16;
        const float bb = bo1[col];
#pragma unroll
        for (int v = 0; v < 8; ++v) sAcc[v + half * 8][col] = acc[c][v] + bb;
      }
      __syncthreads();
      {
        const int row = tid >> 4, seg = tid & 15;
        float p = 0.f;
#pragma unroll
        for (int j = 0; j < 16; ++j) p += sAcc[row][seg * 16 + j] * Wo2[seg * 16 + j];
        sRed[row][seg] = p;
      }
      __syncthreads();
      if (tid < ROWS) {
        float s = bo2[0];
#pragma unroll
        for (int q = 0; q < 16; ++q) s += sRed[tid][q];
        out[(r0 + tid) * T_STEPS + step] = tanhf(s);
      }
    }

    if (step < T_STEPS - 1) {
      // ---------------- GRU: gh = hp @ Whh^T + b_hh (16x768) ---------------
      v8f acc[6];
      gemm_k256<6>(&sA[0][0], wWhh, wave, lane, acc);
#pragma unroll
      for (int c = 0; c < 6; ++c) {
        const int col = (wave * 6 + c) * 16 + ln16;
        const float bb = b_hh[col];
#pragma unroll
        for (int v = 0; v < 8; ++v) sGh[v + half * 8][col] = acc[c][v] + bb;
      }
      __syncthreads();
      // elementwise gates in f32; x is a scalar per row (W_ih is 768x1)
      for (int idx = tid; idx < ROWS * HID; idx += 256) {
        const int row = idx >> 8, j = idx & 255;
        const float x  = data[step * BATCH + r0 + row];
        const float rr = 1.f / (1.f + expf(-((x * W_ih[j]       + b_ih[j])       + sGh[row][j])));
        const float zz = 1.f / (1.f + expf(-((x * W_ih[256 + j] + b_ih[256 + j]) + sGh[row][256 + j])));
        const float nn = tanhf((x * W_ih[512 + j] + b_ih[512 + j]) + rr * sGh[row][512 + j]);
        sH[row][j] = (1.f - zz) * nn + zz * sH[row][j];
      }
      __syncthreads();
    } else {
      // final step: emit hp_f
      for (int idx = tid; idx < ROWS * HID; idx += 256) {
        const int row = idx >> 8, j = idx & 255;
        hpf[(r0 + row) * HID + j] = sH[row][j];
      }
    }
  }
}

// ---- one-shot f32 -> bf16 weight conversion into workspace ----------------
// ws layout: [0,64K) W1 | [64K,128K) W2 | [128K,320K) Whh | [320K,384K) Wo1
__global__ __launch_bounds__(256) void cvt_weights_bf16(
    const float* __restrict__ W1, const float* __restrict__ W2,
    const float* __restrict__ Whh, const float* __restrict__ Wo1,
    __bf16* __restrict__ ws)
{
  const int i = blockIdx.x * 256 + threadIdx.x;
  if (i < 65536)        ws[i] = (__bf16)W1[i];
  else if (i < 131072)  ws[i] = (__bf16)W2[i - 65536];
  else if (i < 327680)  ws[i] = (__bf16)Whh[i - 131072];
  else if (i < 393216)  ws[i] = (__bf16)Wo1[i - 327680];
}

extern "C" void kernel_launch(void* const* d_in, const int* in_sizes, int n_in,
                              void* d_out, int out_size, void* d_ws, size_t ws_size,
                              hipStream_t stream) {
  (void)in_sizes; (void)n_in; (void)out_size; (void)ws_size;
  const float* data = (const float*)d_in[0];
  const float* t    = (const float*)d_in[1];
  const float* W1   = (const float*)d_in[2];
  const float* b1   = (const float*)d_in[3];
  const float* W2   = (const float*)d_in[4];
  const float* b2   = (const float*)d_in[5];
  const float* W_ih = (const float*)d_in[6];
  const float* b_ih = (const float*)d_in[7];
  const float* W_hh = (const float*)d_in[8];
  const float* b_hh = (const float*)d_in[9];
  const float* Wo1  = (const float*)d_in[10];
  const float* bo1  = (const float*)d_in[11];
  const float* Wo2  = (const float*)d_in[12];
  const float* bo2  = (const float*)d_in[13];
  const float* h0   = (const float*)d_in[14];

  __bf16* ws = reinterpret_cast<__bf16*>(d_ws);
  float* out = (float*)d_out;                 // (B, T) row-major
  float* hpf = out + BATCH * T_STEPS;         // (B, H) row-major

  cvt_weights_bf16<<<1536, 256, 0, stream>>>(W1, W2, W_hh, Wo1, ws);
  ode_rnn_fused<<<BATCH / ROWS, 256, 0, stream>>>(
      data, t, b1, b2, W_ih, b_ih, b_hh, bo1, Wo2, bo2, h0,
      ws, ws + 65536, ws + 131072, ws + 327680, out, hpf);
}
